// Seq2Seq2Seq_56221121905362
// MI455X (gfx1250) — matrix operand order repaired and checked by reading, not compiled, not run on hardware
//
#include <hip/hip_runtime.h>
#include <stdint.h>

// ---------------------------------------------------------------------------
// Seq2seq 2-layer LSTM (B=1024, H=512, T=120+14+16) on gfx1250 (MI455X).
// Strategy: bf16 V_WMMA_F32_16X16X32_BF16 for all recurrent GEMMs, weights
// pre-packed once into WMMA B-fragment lane layout (hot in 192MB L2 across all
// 300 sequential layer-steps), h-state kept in bf16 ping-pong buffers matching
// the WMMA A-fragment layout, c-state + gate math in fp32.
// ---------------------------------------------------------------------------

typedef __attribute__((ext_vector_type(16))) __bf16 v16bf;
typedef __attribute__((ext_vector_type(8)))  float  v8f;

#define Bn 1024   // batch
#define Hn 512    // hidden
#define G4 2048   // 4*H gates

union Frag16 { v16bf v; uint4 q[2]; };

__device__ __forceinline__ float sigmoidf_(float x) { return 1.0f / (1.0f + __expf(-x)); }
__device__ __forceinline__ float tanhf_(float x)    { return 2.0f / (1.0f + __expf(-2.0f * x)) - 1.0f; }

// One LSTM layer-step: gates = A0 @ W0 (+ A1 @ W1) + bias (+ x_small @ Wih0),
// then c,h update. A* are bf16 [1024x512] row-major; W* are bf16 pre-packed
// per 16x32 (NxK) B-tiles: dst[((nt*16+kt)*32+lane)*16+e], n=nt*16+(lane&15),
// k=kt*32+(lane>>4)*16+e. Wave = one 16(batch) x 16(hidden-unit) tile, all 4
// gates accumulated in 4 f32 fragments.
__global__ void __launch_bounds__(256) lstm_step_kernel(
    const __bf16* __restrict__ A0, const __bf16* __restrict__ W0,
    const __bf16* __restrict__ A1, const __bf16* __restrict__ W1,
    const float*  __restrict__ bias,
    const float*  __restrict__ xsmall, const float* __restrict__ Wih0,
    int xoff, int xstride, int in_dim,
    float* __restrict__ c, __bf16* __restrict__ hout)
{
    const int lane  = threadIdx.x & 31;
    const int tile  = blockIdx.x * 8 + (threadIdx.x >> 5);   // 0..2047
    const int mtile = tile >> 5;                             // 0..63 (batch/16)
    const int jtile = tile & 31;                             // 0..31 (hidden/16)
    const int rowA  = mtile * 16 + (lane & 15);
    const int kb    = (lane >> 4) * 16;                      // A-frag K sub-base

    v8f acc[4] = {};                                         // i, f, g, o

    const __bf16* Aseg[2] = {A0, A1};
    const __bf16* Wseg[2] = {W0, W1};
    for (int seg = 0; seg < 2; ++seg) {                      // uniform branch
        const __bf16* A = Aseg[seg];
        if (A == nullptr) break;
        const __bf16* W = Wseg[seg];
        const __bf16* arow = A + (size_t)rowA * Hn;
        for (int kt = 0; kt < Hn / 32; ++kt) {
            Frag16 a;
            // A operand, this code's k order: lanes hold M=lane%16; K chunk
            // [kb..kb+15] (matches packed B) -> two contiguous 16B loads.
            a.q[0] = *(const uint4*)(arow + kt * 32 + kb);
            a.q[1] = *(const uint4*)(arow + kt * 32 + kb + 8);
            __builtin_prefetch(arow + kt * 32 + 32, 0, 1);   // next K-slab
#pragma unroll
            for (int gt = 0; gt < 4; ++gt) {
                Frag16 b;
                const __bf16* bp =
                    W + ((size_t)((gt * 32 + jtile) * 16 + kt) * 32 + lane) * 16;
                b.q[0] = *(const uint4*)(bp);
                b.q[1] = *(const uint4*)(bp + 8);
                acc[gt] = __builtin_amdgcn_wmma_f32_16x16x32_bf16(
                    false, a.v, false, b.v, (short)0, acc[gt], false, false);
            }
        }
    }

    const int j  = jtile * 16 + (lane & 15);
    const float bi = bias[j];
    const float bf = bias[Hn + j];
    const float bg = bias[2 * Hn + j];
    const float bo = bias[3 * Hn + j];

    float wi[4][4];
    if (xsmall != nullptr) {
        for (int gt = 0; gt < 4; ++gt)
            for (int idx = 0; idx < in_dim; ++idx)
                wi[gt][idx] = Wih0[(size_t)(gt * Hn + j) * in_dim + idx];
    }

    const int rbase = (lane >> 4) * 8;   // C/D layout: VGPR r -> M = rbase + r
#pragma unroll
    for (int r = 0; r < 8; ++r) {
        const int m = mtile * 16 + rbase + r;
        float gi = acc[0][r] + bi;
        float gf = acc[1][r] + bf;
        float gg = acc[2][r] + bg;
        float go = acc[3][r] + bo;
        if (xsmall != nullptr) {
            for (int idx = 0; idx < in_dim; ++idx) {
                const float xv = xsmall[(size_t)m * xstride + xoff + idx];
                gi += xv * wi[0][idx];
                gf += xv * wi[1][idx];
                gg += xv * wi[2][idx];
                go += xv * wi[3][idx];
            }
        }
        const size_t cix = (size_t)m * Hn + j;
        const float cn = sigmoidf_(gf) * c[cix] + sigmoidf_(gi) * tanhf_(gg);
        c[cix] = cn;
        hout[cix] = (__bf16)(sigmoidf_(go) * tanhf_(cn));
    }
}

// fp32 [2048x512] row-major -> bf16 packed WMMA B-fragment layout.
__global__ void __launch_bounds__(256) pack_weight_kernel(
    const float* __restrict__ src, __bf16* __restrict__ dst)
{
    const int idx  = blockIdx.x * 256 + threadIdx.x;   // grid exact: 4096*256 = 2048*512
    const int e    = idx & 15;
    const int lane = (idx >> 4) & 31;
    const int kt   = (idx >> 9) & 15;
    const int nt   = idx >> 13;
    const int n    = nt * 16 + (lane & 15);
    const int k    = kt * 32 + (lane >> 4) * 16 + e;
    dst[idx] = (__bf16)src[(size_t)n * Hn + k];
}

__global__ void __launch_bounds__(256) combine_bias_kernel(
    const float* __restrict__ bih, const float* __restrict__ bhh,
    float* __restrict__ dst)
{
    const int i = blockIdx.x * 256 + threadIdx.x;      // grid exact: 8*256 = 2048
    dst[i] = bih[i] + bhh[i];
}

__global__ void __launch_bounds__(256) zero_kernel(uint32_t* __restrict__ p, int n)
{
    const int i = blockIdx.x * 256 + threadIdx.x;
    if (i < n) p[i] = 0u;
}

// y[b, t] = dot(h1[b, :], fcW) + fcb   (one wave per batch row)
__global__ void __launch_bounds__(256) fc_kernel(
    const __bf16* __restrict__ h, const float* __restrict__ W,
    const float* __restrict__ bptr, float* __restrict__ out, int t)
{
    const int wave = (blockIdx.x * blockDim.x + threadIdx.x) >> 5;  // 0..1023 exact
    const int lane = threadIdx.x & 31;
    float s = 0.0f;
    for (int jj = lane; jj < Hn; jj += 32)
        s += (float)h[(size_t)wave * Hn + jj] * W[jj];
    for (int off = 16; off > 0; off >>= 1)
        s += __shfl_down(s, off, 32);
    if (lane == 0) out[(size_t)wave * 30 + t] = s + bptr[0];
}

extern "C" void kernel_launch(void* const* d_in, const int* in_sizes, int n_in,
                              void* d_out, int out_size, void* d_ws, size_t ws_size,
                              hipStream_t stream)
{
    (void)in_sizes; (void)n_in; (void)out_size; (void)ws_size;
    char* ws = (char*)d_ws;

    // ---- workspace layout (~26 MB) ----
    const size_t PKB = (size_t)G4 * Hn * sizeof(__bf16);   // 2 MB per packed W
    __bf16* pk[9];
    for (int i = 0; i < 9; ++i) pk[i] = (__bf16*)(ws + (size_t)i * PKB);
    size_t off = 9 * PKB;
    float* bias = (float*)(ws + off); off += (size_t)6 * G4 * sizeof(float);
    float* c0   = (float*)(ws + off); off += (size_t)Bn * Hn * sizeof(float);
    float* c1   = (float*)(ws + off); off += (size_t)Bn * Hn * sizeof(float);
    __bf16* h0b[2]; __bf16* h1b[2];
    h0b[0] = (__bf16*)(ws + off); off += (size_t)Bn * Hn * sizeof(__bf16);
    h0b[1] = (__bf16*)(ws + off); off += (size_t)Bn * Hn * sizeof(__bf16);
    h1b[0] = (__bf16*)(ws + off); off += (size_t)Bn * Hn * sizeof(__bf16);
    h1b[1] = (__bf16*)(ws + off); off += (size_t)Bn * Hn * sizeof(__bf16);

    // ---- zero initial h/c state (c0..h1b[1] are contiguous: 8 MB) ----
    {
        const int ndw = (2 * Bn * Hn * 4 + 4 * Bn * Hn * 2) / 4;  // 2,097,152
        zero_kernel<<<(ndw + 255) / 256, 256, 0, stream>>>((uint32_t*)c0, ndw);
    }

    // ---- pack the 9 big (2048x512) weight matrices to bf16 WMMA layout ----
    const int wsrc[9] = {4, 7, 8, 12, 15, 16, 20, 23, 24};
    for (int i = 0; i < 9; ++i)
        pack_weight_kernel<<<(G4 * Hn) / 256, 256, 0, stream>>>(
            (const float*)d_in[wsrc[i]], pk[i]);

    // ---- combine bih + bhh for the 6 layer-instances ----
    const int bsrc[6][2] = {{5, 6}, {9, 10}, {13, 14}, {17, 18}, {21, 22}, {25, 26}};
    for (int i = 0; i < 6; ++i)
        combine_bias_kernel<<<G4 / 256, 256, 0, stream>>>(
            (const float*)d_in[bsrc[i][0]], (const float*)d_in[bsrc[i][1]],
            bias + (size_t)i * G4);

    const float* fcW = (const float*)d_in[27];
    const float* fcb = (const float*)d_in[28];
    float* out = (float*)d_out;

    // ---- sequential phases; state carries through shared buffers ----
    int s = 0;  // global step counter -> ping-pong parity
    auto run_phase = [&](const float* x, const float* Wih0raw, int T, int in_dim,
                         __bf16* pWhh0, __bf16* pWih1, __bf16* pWhh1,
                         const float* b0, const float* b1, int fc_t0) {
        for (int t = 0; t < T; ++t) {
            const int cur = s & 1, nxt = cur ^ 1;
            // layer 0: K=512 recurrent + tiny dense input
            lstm_step_kernel<<<256, 256, 0, stream>>>(
                h0b[cur], pWhh0, nullptr, nullptr, b0,
                x, Wih0raw, t * in_dim, T * in_dim, in_dim, c0, h0b[nxt]);
            // layer 1: K=1024 = [h0_new | h1_old]
            lstm_step_kernel<<<256, 256, 0, stream>>>(
                h0b[nxt], pWih1, h1b[cur], pWhh1, b1,
                nullptr, nullptr, 0, 0, 0, c1, h1b[nxt]);
            if (fc_t0 >= 0)
                fc_kernel<<<128, 256, 0, stream>>>(h1b[nxt], fcW, fcb, out, fc_t0 + t);
            ++s;
        }
    };

    run_phase((const float*)d_in[0],  (const float*)d_in[3],  120, 4,
              pk[0], pk[1], pk[2], bias + 0 * G4, bias + 1 * G4, -1);   // encoder
    run_phase((const float*)d_in[1],  (const float*)d_in[11], 14, 3,
              pk[3], pk[4], pk[5], bias + 2 * G4, bias + 3 * G4, 0);    // decoder 1
    run_phase((const float*)d_in[2],  (const float*)d_in[19], 16, 1,
              pk[6], pk[7], pk[8], bias + 4 * G4, bias + 5 * G4, 14);   // decoder 2
}
